// MultiHeadAttentionSolution_43215960932835
// MI455X (gfx1250) — matrix operand order repaired and checked by reading, not compiled, not run on hardware
//
#include <hip/hip_runtime.h>

// ---------------- problem constants ----------------
#define B_SZ    4
#define T_SEQ   2048
#define D_MODEL 1024
#define N_HEADS 16
#define D_KHEAD 64

typedef __attribute__((ext_vector_type(16))) __bf16        bf16x16;
typedef __attribute__((ext_vector_type(8)))  float         f32x8;
typedef __attribute__((ext_vector_type(4)))  unsigned int  u32x4;

union Frag {
    u32x4   u[2];   // two b128 loads
    bf16x16 v;      // WMMA operand
};

__device__ __forceinline__ unsigned short f2bf(float f) {
    unsigned int u = __float_as_uint(f);
    u += 0x7FFFu + ((u >> 16) & 1u);   // round-to-nearest-even
    return (unsigned short)(u >> 16);
}

__device__ __forceinline__ f32x8 wmma_bf16(bf16x16 a, bf16x16 b, f32x8 c) {
    // (neg_a, A, neg_b, B, c_mod, C, reuse_a, reuse_b)
    return __builtin_amdgcn_wmma_f32_16x16x32_bf16(false, a, false, b, (short)0, c, false, false);
}

// ---------------- f32 -> bf16 convert ----------------
__global__ void cvt_f32_bf16(const float* __restrict__ in,
                             unsigned short* __restrict__ out, int n) {
    int i = blockIdx.x * blockDim.x + threadIdx.x;
    if (i < n) out[i] = f2bf(in[i]);
}

// ---------------- NT GEMM: Y[m,n] = sum_k A[m,k]*Bw[n,k] ----------------
// A: MxK bf16 row-major, Bw: NxK bf16 row-major (both K-contiguous).
// mode 0: bf16 out, Y[m*N+n]
// mode 1: bf16 out, transposed per batch: Y[(b*N+n)*Tdim + t], m = b*Tdim + t
// mode 2: f32  out, Y[m*N+n]
__global__ __launch_bounds__(128) void gemm_nt_bf16(
    const unsigned short* __restrict__ A,
    const unsigned short* __restrict__ Bw,
    void* __restrict__ Y,
    int M, int N, int K, int mode, int Tdim)
{
    const int lane  = threadIdx.x & 31;
    const int wave  = threadIdx.x >> 5;           // 4 waves, 2x2 of 64x64
    const int l16   = lane & 15;
    const int lhalf = lane >> 4;                  // 0 or 1

    const int m0 = blockIdx.x * 128 + (wave >> 1) * 64;
    const int n0 = blockIdx.y * 128 + (wave & 1) * 64;

    f32x8 acc[4][4];
#pragma unroll
    for (int i = 0; i < 4; ++i)
#pragma unroll
        for (int j = 0; j < 4; ++j)
#pragma unroll
            for (int r = 0; r < 8; ++r) acc[i][j][r] = 0.0f;

    for (int kb = 0; kb < K; kb += 32) {
        Frag a[4], b[4];
#pragma unroll
        for (int i = 0; i < 4; ++i) {
            // A layout: lane = row, 16 contiguous K values (lhalf selects K 0-15 / 16-31)
            const unsigned short* ap =
                A + (size_t)(m0 + i * 16 + l16) * K + kb + lhalf * 16;
            a[i].u[0] = *(const u32x4*)ap;
            a[i].u[1] = *(const u32x4*)(ap + 8);
        }
#pragma unroll
        for (int j = 0; j < 4; ++j) {
            // B layout: lane = column, 16 contiguous K values (lhalf selects K 0-15 / 16-31)
            const unsigned short* bp =
                Bw + (size_t)(n0 + j * 16 + l16) * K + kb + lhalf * 16;
            b[j].u[0] = *(const u32x4*)bp;
            b[j].u[1] = *(const u32x4*)(bp + 8);
        }
#pragma unroll
        for (int i = 0; i < 4; ++i)
#pragma unroll
            for (int j = 0; j < 4; ++j)
                acc[i][j] = wmma_bf16(a[i].v, b[j].v, acc[i][j]);
    }

    // C layout: VGPR r -> row (r + 8*lhalf), lane%16 -> column
#pragma unroll
    for (int i = 0; i < 4; ++i)
#pragma unroll
        for (int j = 0; j < 4; ++j) {
            const int n = n0 + j * 16 + l16;
#pragma unroll
            for (int r = 0; r < 8; ++r) {
                const int m = m0 + i * 16 + r + lhalf * 8;
                const float v = acc[i][j][r];
                if (mode == 2) {
                    ((float*)Y)[(size_t)m * N + n] = v;
                } else {
                    const unsigned short h = f2bf(v);
                    if (mode == 0) {
                        ((unsigned short*)Y)[(size_t)m * N + n] = h;
                    } else {
                        const int bb = m / Tdim;
                        const int t  = m - bb * Tdim;
                        ((unsigned short*)Y)[((size_t)bb * N + n) * Tdim + t] = h;
                    }
                }
            }
        }
}

// ---------------- causal flash attention ----------------
// Q,K: (B*T, D_MODEL) bf16 row-major; Vt: (B, D_MODEL, T) bf16 (t contiguous)
// ctx: (B*T, D_MODEL) bf16. One wave = one 16-query tile of one (b,h).
__global__ __launch_bounds__(128) void flash_attn_bf16(
    const unsigned short* __restrict__ Qb,
    const unsigned short* __restrict__ Kb,
    const unsigned short* __restrict__ Vt,
    unsigned short* __restrict__ ctx)
{
    __shared__ unsigned short plds[4][16 * 32];   // per-wave P staging (C->A relayout)

    const int lane  = threadIdx.x & 31;
    const int wave  = threadIdx.x >> 5;
    const int l16   = lane & 15;
    const int lhalf = lane >> 4;

    const int tile = blockIdx.x * 4 + wave;
    const int tiles_per_bh = T_SEQ / 16;
    const int bh = tile / tiles_per_bh;
    const int q0 = (tile % tiles_per_bh) * 16;
    const int b  = bh / N_HEADS;
    const int h  = bh % N_HEADS;

    const unsigned short* Qp = Qb + (size_t)b * T_SEQ * D_MODEL + h * D_KHEAD;
    const unsigned short* Kp = Kb + (size_t)b * T_SEQ * D_MODEL + h * D_KHEAD;
    const unsigned short* Vp = Vt + ((size_t)b * D_MODEL + h * D_KHEAD) * T_SEQ;

    // Q A-fragments for dk chunks 0..31 and 32..63 (loaded once)
    Frag qa[2];
#pragma unroll
    for (int f = 0; f < 2; ++f) {
        const unsigned short* p = Qp + (size_t)(q0 + l16) * D_MODEL + f * 32 + lhalf * 16;
        qa[f].u[0] = *(const u32x4*)p;
        qa[f].u[1] = *(const u32x4*)(p + 8);
    }

    float mrow[8], lrow[8];
#pragma unroll
    for (int r = 0; r < 8; ++r) { mrow[r] = -1e30f; lrow[r] = 0.0f; }
    f32x8 oacc[4];
#pragma unroll
    for (int j = 0; j < 4; ++j)
#pragma unroll
        for (int r = 0; r < 8; ++r) oacc[j][r] = 0.0f;

    const float sm_scale = 0.125f;       // 1/sqrt(64)
    const float L2E = 1.44269504088896f;

    for (int n0 = 0; n0 < q0 + 16; n0 += 32) {        // causal: keys <= q0+15
        // ---- S = Q K^T (16 x 32), two 16x16 C tiles ----
        f32x8 sacc[2];
#pragma unroll
        for (int s = 0; s < 2; ++s)
#pragma unroll
            for (int r = 0; r < 8; ++r) sacc[s][r] = 0.0f;
#pragma unroll
        for (int s = 0; s < 2; ++s)
#pragma unroll
            for (int f = 0; f < 2; ++f) {
                Frag kf;
                const unsigned short* p =
                    Kp + (size_t)(n0 + s * 16 + l16) * D_MODEL + f * 32 + lhalf * 16;
                kf.u[0] = *(const u32x4*)p;
                kf.u[1] = *(const u32x4*)(p + 8);
                sacc[s] = wmma_bf16(qa[f].v, kf.v, sacc[s]);
            }

        // ---- online softmax (row = r + 8*lhalf across lanes l16) ----
        float pe0[8], pe1[8];
#pragma unroll
        for (int r = 0; r < 8; ++r) {
            const int qrow = q0 + r + lhalf * 8;
            float s0 = sacc[0][r] * sm_scale;
            float s1 = sacc[1][r] * sm_scale;
            if (n0 + l16 > qrow)      s0 = -1e30f;
            if (n0 + 16 + l16 > qrow) s1 = -1e30f;
            float mx = fmaxf(s0, s1);
#pragma unroll
            for (int off = 8; off; off >>= 1)
                mx = fmaxf(mx, __shfl_xor(mx, off, 16));
            const float mnew = fmaxf(mrow[r], mx);
            const float corr = exp2f((mrow[r] - mnew) * L2E);
            const float e0 = exp2f((s0 - mnew) * L2E);
            const float e1 = exp2f((s1 - mnew) * L2E);
            float rs = e0 + e1;
#pragma unroll
            for (int off = 8; off; off >>= 1)
                rs += __shfl_xor(rs, off, 16);
            lrow[r] = lrow[r] * corr + rs;
            mrow[r] = mnew;
#pragma unroll
            for (int j = 0; j < 4; ++j) oacc[j][r] *= corr;
            pe0[r] = e0; pe1[r] = e1;
        }

        // ---- relayout P: C-layout -> A-layout via wave-private LDS ----
        unsigned short* pl = plds[wave];
#pragma unroll
        for (int r = 0; r < 8; ++r) {
            const int row = r + lhalf * 8;
            pl[row * 32 + l16]      = f2bf(pe0[r]);
            pl[row * 32 + 16 + l16] = f2bf(pe1[r]);
        }
        asm volatile("s_wait_dscnt 0x0" ::: "memory");   // same-wave DS order
        Frag pa;
        {
            const unsigned short* p = pl + l16 * 32 + lhalf * 16;
            pa.u[0] = *(const u32x4*)p;
            pa.u[1] = *(const u32x4*)(p + 8);
        }

        // ---- O += P V  (V B-frags are key-contiguous via Vt) ----
#pragma unroll
        for (int j = 0; j < 4; ++j) {
            Frag vf;
            const unsigned short* p =
                Vp + (size_t)(j * 16 + l16) * T_SEQ + n0 + lhalf * 16;
            vf.u[0] = *(const u32x4*)p;
            vf.u[1] = *(const u32x4*)(p + 8);
            oacc[j] = wmma_bf16(pa.v, vf.v, oacc[j]);
        }
    }

    // ---- finalize: divide by row sum, store ctx (B*T, D_MODEL) bf16 ----
#pragma unroll
    for (int j = 0; j < 4; ++j) {
        const int d = h * D_KHEAD + j * 16 + l16;
#pragma unroll
        for (int r = 0; r < 8; ++r) {
            const int qrow = q0 + r + lhalf * 8;
            const float v = oacc[j][r] / lrow[r];
            ctx[(size_t)(b * T_SEQ + qrow) * D_MODEL + d] = f2bf(v);
        }
    }
}

// ---------------- launcher ----------------
extern "C" void kernel_launch(void* const* d_in, const int* in_sizes, int n_in,
                              void* d_out, int out_size, void* d_ws, size_t ws_size,
                              hipStream_t stream) {
    (void)in_sizes; (void)n_in; (void)out_size; (void)ws_size;

    const float* x  = (const float*)d_in[0];
    const float* Wq = (const float*)d_in[1];
    const float* Wk = (const float*)d_in[2];
    const float* Wv = (const float*)d_in[3];
    const float* Wo = (const float*)d_in[4];
    float* out = (float*)d_out;

    const int M = B_SZ * T_SEQ;          // 8192
    const int D = D_MODEL;               // 1024
    const int xN = M * D;                // 8,388,608
    const int wN = D * D;                // 1,048,576

    unsigned short* ws  = (unsigned short*)d_ws;
    size_t off = 0;
    unsigned short* xb  = ws + off; off += (size_t)xN;   // x bf16 (reused as ctx later)
    unsigned short* wqb = ws + off; off += (size_t)wN;
    unsigned short* wkb = ws + off; off += (size_t)wN;
    unsigned short* wvb = ws + off; off += (size_t)wN;
    unsigned short* wob = ws + off; off += (size_t)wN;
    unsigned short* Qb  = ws + off; off += (size_t)xN;
    unsigned short* Kb  = ws + off; off += (size_t)xN;
    unsigned short* Vtb = ws + off; off += (size_t)xN;
    unsigned short* ctx = xb;            // alias: x dead after projections

    // 1) downconvert
    cvt_f32_bf16<<<(xN + 255) / 256, 256, 0, stream>>>(x,  xb,  xN);
    cvt_f32_bf16<<<(wN + 255) / 256, 256, 0, stream>>>(Wq, wqb, wN);
    cvt_f32_bf16<<<(wN + 255) / 256, 256, 0, stream>>>(Wk, wkb, wN);
    cvt_f32_bf16<<<(wN + 255) / 256, 256, 0, stream>>>(Wv, wvb, wN);
    cvt_f32_bf16<<<(wN + 255) / 256, 256, 0, stream>>>(Wo, wob, wN);

    // 2) projections (y = x W^T); V stored transposed per batch for PV GEMM
    dim3 gg(M / 128, D / 128);           // (64, 8)
    gemm_nt_bf16<<<gg, 128, 0, stream>>>(xb, wqb, Qb,  M, D, D, 0, T_SEQ);
    gemm_nt_bf16<<<gg, 128, 0, stream>>>(xb, wkb, Kb,  M, D, D, 0, T_SEQ);
    gemm_nt_bf16<<<gg, 128, 0, stream>>>(xb, wvb, Vtb, M, D, D, 1, T_SEQ);

    // 3) causal flash attention: 512 q-tiles, 4 waves/block
    flash_attn_bf16<<<(M / 16) / 4, 128, 0, stream>>>(Qb, Kb, Vtb, ctx);

    // 4) output projection, f32 result
    gemm_nt_bf16<<<gg, 128, 0, stream>>>(ctx, wob, out, M, D, D, 2, T_SEQ);
}
